// MultiHeadAttention_4380866642010
// MI455X (gfx1250) — matrix operand order repaired and checked by reading, not compiled, not run on hardware
//
#include <hip/hip_runtime.h>
#include <hip/hip_bf16.h>

#define DEV __device__ __forceinline__

typedef unsigned short u16;
typedef __attribute__((ext_vector_type(4)))  u16     u16x4;
typedef __attribute__((ext_vector_type(8)))  u16     u16x8;
typedef __attribute__((ext_vector_type(16))) u16     u16x16;
typedef __attribute__((ext_vector_type(4)))  float   f32x4;
typedef __attribute__((ext_vector_type(16))) __bf16  v16bf;
typedef __attribute__((ext_vector_type(8)))  float   v8f;

static constexpr int NE    = 1024;    // embed dim C
static constexpr int NH    = 16;      // heads
static constexpr int HD    = 64;      // head dim
static constexpr int BB    = 2;       // batch
static constexpr int TT    = 2048;    // seq len
static constexpr int MROWS = BB * TT; // 4096

// scale * log2(e): softmax evaluated in exp2 domain (one v_exp_f32 per exp)
#define SM_SCALE_LOG2 (0.125f * 1.44269504088896340736f)
#define NEG_BIG (-1e30f)

// ---- helpers -------------------------------------------------------------

DEV u16 f2bf(float f) {
  unsigned u = __builtin_bit_cast(unsigned, f);
  u += 0x7FFFu + ((u >> 16) & 1u);        // round-to-nearest-even
  return (u16)(u >> 16);
}

// A-fragment for 16x16x32 bf16 WMMA: lane holds two contiguous 8-element runs.
DEV v16bf frag2(const u16* lo, const u16* hi) {
  u16x8 a = *(const u16x8*)lo;
  u16x8 b = *(const u16x8*)hi;
  u16x16 c;
#pragma unroll
  for (int i = 0; i < 8; i++) { c[i] = a[i]; c[i + 8] = b[i]; }
  return __builtin_bit_cast(v16bf, c);
}

// B-fragment: lane holds 16 contiguous K elements (data must be N-major).
DEV v16bf frag16(const u16* p) {
  return __builtin_bit_cast(v16bf, *(const u16x16*)p);
}

DEV v8f wmma_bf16(v16bf a, v16bf b, v8f c) {
  return __builtin_amdgcn_wmma_f32_16x16x32_bf16(
      /*neg_a=*/false, a, /*neg_b=*/false, b,
      /*c_mod=*/(short)0, c, /*reuse_a=*/false, /*reuse_b=*/false);
}

// ---- prep kernels --------------------------------------------------------

// vectorized fp32 -> bf16 (float4 in, 4x u16 out)
__global__ void cvt_kernel(const float* __restrict__ src, u16* __restrict__ dst, int n4) {
  int i = blockIdx.x * blockDim.x + threadIdx.x;
  if (i < n4) {
    f32x4 v = ((const f32x4*)src)[i];
    u16x4 o;
#pragma unroll
    for (int j = 0; j < 4; j++) o[j] = f2bf(v[j]);
    ((u16x4*)dst)[i] = o;
  }
}

// src fp32 [K][N] row-major -> dst bf16 [N][K] row-major (B^T), LDS-tiled so
// both the global read (over n) and the global write (over k) are coalesced.
__global__ void tr_cvt_kernel(const float* __restrict__ src, u16* __restrict__ dst,
                              int K, int N) {
  __shared__ u16 tile[32][33];              // +1 pad: no bank conflicts
  const int kb = blockIdx.x * 32;
  const int nb = blockIdx.y * 32;
  const int tx = threadIdx.x & 31;
  const int ty = threadIdx.x >> 5;          // 256 threads -> 8 rows per pass
#pragma unroll
  for (int r = ty; r < 32; r += 8)
    tile[r][tx] = f2bf(src[(size_t)(kb + r) * N + nb + tx]);   // tile[k][n]
  __syncthreads();
#pragma unroll
  for (int r = ty; r < 32; r += 8)
    dst[(size_t)(nb + r) * K + kb + tx] = tile[tx][r];          // dst[n][k]
}

// ---- GEMM: C = A(bf16, MxK) * B(bf16, KxN given as BT NxK) + bias --------
// 128 threads = 4 waves; each wave a 32x32 tile (2x2 WMMA tiles, fragments
// reused across the 2x2 -> half the vmem traffic per FLOP); block = 64x64.
// MODE 0: write fp32 out[M][N].
// MODE 1: QKV scatter -> Qb[B,H,T,D], Kb[B,H,T,D], VTb[B,H,D,T] (bf16).

template <int MODE>
DEV void gemm_epilogue(v8f acc, int m0, int n0, int half, int l16,
                       const float* __restrict__ bias, float* __restrict__ outF,
                       u16* __restrict__ Qb, u16* __restrict__ Kb,
                       u16* __restrict__ VTb, int N) {
  const int   n  = n0 + l16;
  const float bv = bias[n];
#pragma unroll
  for (int r = 0; r < 8; r++) {
    const int   m = m0 + r + half * 8;
    const float v = acc[r] + bv;
    if (MODE == 0) {
      outF[(size_t)m * N + n] = v;
    } else {
      const int sec = n >> 10;       // which of Q/K/V (N = 3*1024)
      const int c   = n & 1023;
      const int h   = c >> 6;
      const int d   = c & 63;
      const int b   = m >> 11;       // T = 2048
      const int t   = m & 2047;
      const size_t bh = (size_t)(b * NH + h);
      const u16 w = f2bf(v);
      if (sec == 0)      Qb [(bh * TT + t) * HD + d] = w;
      else if (sec == 1) Kb [(bh * TT + t) * HD + d] = w;
      else               VTb[(bh * HD + d) * TT + t] = w;   // V transposed
    }
  }
}

template <int MODE>
__global__ __launch_bounds__(128, 1)
void gemm_bf16_kernel(const u16* __restrict__ A, const u16* __restrict__ BT,
                      const float* __restrict__ bias,
                      float* __restrict__ outF,
                      u16* __restrict__ Qb, u16* __restrict__ Kb,
                      u16* __restrict__ VTb,
                      int M, int N, int K) {
  const int lane = threadIdx.x & 31;
  const int wid  = threadIdx.x >> 5;
  const int m0   = blockIdx.y * 64 + (wid >> 1) * 32;
  const int n0   = blockIdx.x * 64 + (wid & 1) * 32;
  const int half = lane >> 4, l16 = lane & 15;
  const int kbA  = half * 16;    // A-fragment lane K base
  const int kbB  = half * 16;    // B-fragment lane K base

  const u16* Arow0 = A  + (size_t)(m0 + l16) * K;
  const u16* Arow1 = A  + (size_t)(m0 + 16 + l16) * K;
  const u16* Brow0 = BT + (size_t)(n0 + l16) * K;
  const u16* Brow1 = BT + (size_t)(n0 + 16 + l16) * K;

  const v8f zero = {0.f, 0.f, 0.f, 0.f, 0.f, 0.f, 0.f, 0.f};
  v8f acc00 = zero, acc01 = zero, acc10 = zero, acc11 = zero;

#pragma unroll 2
  for (int k = 0; k < K; k += 32) {
    v16bf a0 = frag2(Arow0 + k + kbA, Arow0 + k + 8 + kbA);
    v16bf a1 = frag2(Arow1 + k + kbA, Arow1 + k + 8 + kbA);
    v16bf b0 = frag16(Brow0 + k + kbB);
    v16bf b1 = frag16(Brow1 + k + kbB);
    acc00 = wmma_bf16(a0, b0, acc00);
    acc01 = wmma_bf16(a0, b1, acc01);
    acc10 = wmma_bf16(a1, b0, acc10);
    acc11 = wmma_bf16(a1, b1, acc11);
  }

  gemm_epilogue<MODE>(acc00, m0,      n0,      half, l16, bias, outF, Qb, Kb, VTb, N);
  gemm_epilogue<MODE>(acc01, m0,      n0 + 16, half, l16, bias, outF, Qb, Kb, VTb, N);
  gemm_epilogue<MODE>(acc10, m0 + 16, n0,      half, l16, bias, outF, Qb, Kb, VTb, N);
  gemm_epilogue<MODE>(acc11, m0 + 16, n0 + 16, half, l16, bias, outF, Qb, Kb, VTb, N);
}

// ---- flash attention: one 16-query tile per wave, 4 waves/block ----------

__global__ __launch_bounds__(128, 1)
void attn_kernel(const u16* __restrict__ Qb, const u16* __restrict__ Kb,
                 const u16* __restrict__ VTb, u16* __restrict__ Ob) {
  __shared__ u16 pstage[4][16 * 32];   // per-wave P staging (C-frag -> A-frag)

  const int lane = threadIdx.x & 31;
  const int wid  = threadIdx.x >> 5;
  const int half = lane >> 4, l16 = lane & 15;
  const int kbA  = half * 16, kbB = half * 16;

  const int bh = blockIdx.y;
  const int b  = bh >> 4;
  const int h  = bh & 15;
  const int q0 = blockIdx.x * 64 + wid * 16;

  const u16* Q  = Qb  + (size_t)bh * TT * HD;
  const u16* Km = Kb  + (size_t)bh * TT * HD;
  const u16* VT = VTb + (size_t)bh * HD * TT;

  // Q A-fragments for d-chunks [0,32) and [32,64) — persistent across key loop
  const u16* qrow = Q + (size_t)(q0 + l16) * HD;
  const v16bf aQ0 = frag2(qrow + kbA,      qrow + 8 + kbA);
  const v16bf aQ1 = frag2(qrow + 32 + kbA, qrow + 40 + kbA);

  const v8f zero = {0.f, 0.f, 0.f, 0.f, 0.f, 0.f, 0.f, 0.f};
  v8f acc[4] = {zero, zero, zero, zero};     // 16 x 64 output accumulator
  float mrow[8], lrow[8];                    // row stats in log2 domain
#pragma unroll
  for (int r = 0; r < 8; r++) { mrow[r] = NEG_BIG; lrow[r] = 0.f; }

  // block-uniform key-tile count (causal); extra tiles mask to -big -> p=0
  const int nTiles = (blockIdx.x * 64 + 63) / 32 + 1;
  u16* ps = pstage[wid];

  for (int j = 0; j < nTiles; j++) {
    const int kbase = j * 32;

    // S = Q K^T for two 16-key sub-tiles (K row-major: contiguous in d)
    float s0[8], s1[8];
#pragma unroll
    for (int t = 0; t < 2; t++) {
      const u16* krow = Km + (size_t)(kbase + t * 16 + l16) * HD;
      v8f sv = zero;
      sv = wmma_bf16(aQ0, frag16(krow + kbB), sv);
      sv = wmma_bf16(aQ1, frag16(krow + 32 + kbB), sv);
      const int col = kbase + t * 16 + l16;
      float* sd = t ? s1 : s0;
#pragma unroll
      for (int r = 0; r < 8; r++) {
        const int row = q0 + r + half * 8;
        const float x = sv[r] * SM_SCALE_LOG2;   // 1/sqrt(64) * log2(e)
        sd[r] = (col > row) ? NEG_BIG : x;       // causal mask
      }
    }

    // online softmax (exp2 domain): row stats across the 16 lanes of a row
#pragma unroll
    for (int r = 0; r < 8; r++) {
      float mx = fmaxf(s0[r], s1[r]);
      mx = fmaxf(mx, __shfl_xor(mx, 8, 32));
      mx = fmaxf(mx, __shfl_xor(mx, 4, 32));
      mx = fmaxf(mx, __shfl_xor(mx, 2, 32));
      mx = fmaxf(mx, __shfl_xor(mx, 1, 32));
      const float mnew  = fmaxf(mrow[r], mx);
      const float alpha = exp2f(mrow[r] - mnew);
      mrow[r] = mnew;
      const float p0 = exp2f(s0[r] - mnew);
      const float p1 = exp2f(s1[r] - mnew);
      s0[r] = p0; s1[r] = p1;
      float psum = p0 + p1;
      psum += __shfl_xor(psum, 8, 32);
      psum += __shfl_xor(psum, 4, 32);
      psum += __shfl_xor(psum, 2, 32);
      psum += __shfl_xor(psum, 1, 32);
      lrow[r] = lrow[r] * alpha + psum;
      acc[0][r] *= alpha; acc[1][r] *= alpha;
      acc[2][r] *= alpha; acc[3][r] *= alpha;
    }

    // stage P (C-fragment layout) into LDS as row-major 16x32 bf16
#pragma unroll
    for (int r = 0; r < 8; r++) {
      const int row = r + half * 8;
      ps[row * 32 + l16]      = f2bf(s0[r]);
      ps[row * 32 + 16 + l16] = f2bf(s1[r]);
    }
    __syncthreads();
    const v16bf aP = frag2(ps + l16 * 32 + kbA, ps + l16 * 32 + 8 + kbA);
    __syncthreads();

    // O += P * V  (V^T row-major: contiguous over keys)
#pragma unroll
    for (int dt = 0; dt < 4; dt++) {
      const u16* vrow = VT + (size_t)(dt * 16 + l16) * TT + kbase;
      acc[dt] = wmma_bf16(aP, frag16(vrow + kbB), acc[dt]);
    }
  }

  // epilogue: divide by row sums, write bf16 attn-out [B,T,C]
#pragma unroll
  for (int dt = 0; dt < 4; dt++) {
#pragma unroll
    for (int r = 0; r < 8; r++) {
      const int   row = q0 + r + half * 8;
      const float v   = acc[dt][r] / lrow[r];
      Ob[((size_t)b * TT + row) * NE + h * HD + dt * 16 + l16] = f2bf(v);
    }
  }
}

// ---- host launch ---------------------------------------------------------

extern "C" void kernel_launch(void* const* d_in, const int* in_sizes, int n_in,
                              void* d_out, int out_size, void* d_ws, size_t ws_size,
                              hipStream_t stream) {
  (void)in_sizes; (void)n_in; (void)out_size; (void)ws_size;

  const float* x      = (const float*)d_in[0];
  const float* W_attn = (const float*)d_in[1];
  const float* b_attn = (const float*)d_in[2];
  const float* W_out  = (const float*)d_in[3];
  const float* b_out  = (const float*)d_in[4];
  float* out = (float*)d_out;

  // workspace carve-up (all chunks multiples of 2 MB -> aligned)
  char* ws = (char*)d_ws;
  u16* xb  = (u16*)ws; ws += (size_t)MROWS * NE * 2;         // 8 MB
  u16* WaT = (u16*)ws; ws += (size_t)3 * NE * NE * 2;        // 6 MB
  u16* WoT = (u16*)ws; ws += (size_t)NE * NE * 2;            // 2 MB
  u16* Qb  = (u16*)ws; ws += (size_t)BB * NH * TT * HD * 2;  // 8 MB
  u16* Kb  = (u16*)ws; ws += (size_t)BB * NH * TT * HD * 2;  // 8 MB
  u16* VTb = (u16*)ws; ws += (size_t)BB * NH * HD * TT * 2;  // 8 MB
  u16* Ob  = (u16*)ws;                                       // 8 MB

  // prep: fp32 -> bf16 (vectorized), weights transposed via LDS tiles
  const int n4 = MROWS * NE / 4;
  cvt_kernel<<<(n4 + 255) / 256, 256, 0, stream>>>(x, xb, n4);
  tr_cvt_kernel<<<dim3(NE / 32, 3 * NE / 32), 256, 0, stream>>>(W_attn, WaT, NE, 3 * NE);
  tr_cvt_kernel<<<dim3(NE / 32, NE / 32),     256, 0, stream>>>(W_out,  WoT, NE, NE);

  // QKV projection (WMMA, 2x2 tiles/wave) with scatter into Q/K/V^T layouts
  dim3 g1(3 * NE / 64, MROWS / 64);
  gemm_bf16_kernel<1><<<g1, 128, 0, stream>>>(xb, WaT, b_attn, nullptr,
                                              Qb, Kb, VTb, MROWS, 3 * NE, NE);

  // flash attention (WMMA + LDS re-layout + half-wave shuffle softmax)
  dim3 g2(TT / 64, BB * NH);
  attn_kernel<<<g2, 128, 0, stream>>>(Qb, Kb, VTb, Ob);

  // output projection (WMMA) -> fp32 d_out
  dim3 g3(NE / 64, MROWS / 64);
  gemm_bf16_kernel<0><<<g3, 128, 0, stream>>>(Ob, WoT, b_out, out,
                                              nullptr, nullptr, nullptr,
                                              MROWS, NE, NE);
}
